// MRGCN_60155311948054
// MI455X (gfx1250) — matrix operand order repaired and checked by reading, not compiled, not run on hardware
//
#include <hip/hip_runtime.h>
#include <hip/hip_bf16.h>
#include <math.h>

// ---- problem constants (match reference) ----
static constexpr int GG   = 40;       // graphs
static constexpr int NPG  = 500;      // nodes per graph
static constexpr int NN   = 20000;    // total nodes (= 1250 * 16)
static constexpr int EE   = 320000;   // total edges
static constexpr int RR   = 8;        // relations
static constexpr int HID  = 256;
static constexpr int TOT  = 768;      // 3 * HID
static constexpr int KK   = 250;      // top-k kept per graph
static constexpr int LSTM_H = 256;

typedef __attribute__((ext_vector_type(16))) _Float16 v16h;
typedef __attribute__((ext_vector_type(8)))  float    v8f;

// ---------------------------------------------------------------------------
// 1) per-(dst,relation) edge counts for mean normalization
__global__ void k_count(const int* __restrict__ dst, const int* __restrict__ et,
                        int* __restrict__ cnt) {
  int e = blockIdx.x * blockDim.x + threadIdx.x;
  if (e < EE) atomicAdd(&cnt[dst[e] * RR + et[e]], 1);
}

// ---------------------------------------------------------------------------
// 2) convert conv weights to f16, transposed [out][k] for contiguous B loads.
//    Layout: Wt[(l*9+rb)][o][k]; rb in 0..7 = relation, rb==8 = root.
__global__ void k_wcvt(const float* __restrict__ convW,
                       const float* __restrict__ convRoot,
                       _Float16* __restrict__ Wt) {
  int idx = blockIdx.x * blockDim.x + threadIdx.x;
  if (idx >= 3 * 9 * 65536) return;
  int lrb = idx >> 16;
  int rem = idx & 65535;
  int o = rem >> 8, k = rem & 255;
  int l = lrb / 9, rb = lrb % 9;
  float v;
  if (rb < 8) v = convW[((size_t)(l * 8 + rb) * 256 + k) * 256 + o];
  else        v = convRoot[((size_t)l * 256 + k) * 256 + o];
  Wt[((size_t)lrb << 16) + (size_t)o * 256 + k] = (_Float16)v;
}

// ---------------------------------------------------------------------------
// 3) streaming f32 -> f16 matrix convert (4 elems/thread, packs to cvt_pk)
__global__ void k_cvt(const float* __restrict__ in, _Float16* __restrict__ out,
                      int n) {
  int i = (blockIdx.x * blockDim.x + threadIdx.x) * 4;
  if (i + 3 < n) {
    float4 v = *(const float4*)(in + i);
    out[i]     = (_Float16)v.x;
    out[i + 1] = (_Float16)v.y;
    out[i + 2] = (_Float16)v.z;
    out[i + 3] = (_Float16)v.w;
  } else {
    for (; i < n; ++i) out[i] = (_Float16)in[i];
  }
}

// ---------------------------------------------------------------------------
// 4) per-relation mean aggregation: buf[dst] += norm * h[src]  (one wave/edge)
__global__ void k_scatter(const int* __restrict__ src, const int* __restrict__ dst,
                          const int* __restrict__ et, const int* __restrict__ cnt,
                          const float* __restrict__ hin, float* __restrict__ buf,
                          int rel) {
  int gw   = (blockIdx.x * blockDim.x + threadIdx.x) >> 5;
  int lane = threadIdx.x & 31;
  if (gw >= EE) return;
  if (et[gw] != rel) return;
  int s = src[gw], d = dst[gw];
  float w = 1.0f / fmaxf((float)cnt[d * RR + rel], 1.0f);
  const float* xs = hin + (size_t)s * HID;
  float*       bd = buf + (size_t)d * HID;
  #pragma unroll
  for (int i = lane; i < HID; i += 32) atomicAdd(&bd[i], w * xs[i]);
}

// ---------------------------------------------------------------------------
// 5) WMMA GEMM: C[1250*16, 256] (+)= A16[.,256] @ W(256x256, f16 transposed).
//    A is pre-converted f16 -> fragments load directly via global_load_b128,
//    no LDS, no barriers: 8 independent K-steps the scheduler can overlap.
//    Block = 512 threads = 16 waves; each wave owns one 16x16 C tile of the
//    shared M-tile (A rows reused through WGP$).
//    mode bit0: init acc; bit1: finalize (bias+relu, write C, xc slice, and
//    f16 copy of the layer output for the next layer's root pass).
__global__ void k_gemm16(const _Float16* __restrict__ A16,  // [NN][256]
                         const _Float16* __restrict__ Wt,   // [256 out][256 k]
                         float* __restrict__ C,
                         const float* __restrict__ bias,
                         float* __restrict__ xc, int xccol,
                         _Float16* __restrict__ h16out,
                         int mode) {
  const int tid  = threadIdx.x;
  const int lane = tid & 31;
  const int wv   = tid >> 5;                 // wave 0..15 (wave32)
  const int m    = lane & 15;
  const int hs   = lane >> 4;                // half-wave select
  const int row0 = blockIdx.x * 16;
  const int col0 = wv * 16;

  // per-lane base pointers
  const _Float16* pA = A16 + (size_t)(row0 + m) * 256;        // A row (f16)
  const _Float16* pB = Wt + (size_t)(col0 + m) * 256;

  v8f acc;
  if (mode & 1) {
    #pragma unroll
    for (int v = 0; v < 8; ++v) acc[v] = 0.0f;
  } else {
    #pragma unroll
    for (int v = 0; v < 8; ++v)
      acc[v] = C[(size_t)(row0 + v + 8 * hs) * HID + col0 + m];
  }

  #pragma unroll
  for (int k0 = 0; k0 < 256; k0 += 32) {
    v16h af, bf;
    // A fragment: two contiguous 8-half groups (16B global_load_b128 each)
    #pragma unroll
    for (int j = 0; j < 8; ++j) {
      af[j]     = pA[k0 + hs * 8 + j];
      af[8 + j] = pA[k0 + 16 + hs * 8 + j];
    }
    // B fragment: two contiguous 8-half groups (two 16B loads), same K order as A
    if (k0 + 32 < 256) __builtin_prefetch(pB + k0 + 32, 0, 1); // global_prefetch
    #pragma unroll
    for (int j = 0; j < 8; ++j) {
      bf[j]     = pB[k0 + hs * 8 + j];
      bf[8 + j] = pB[k0 + 16 + hs * 8 + j];
    }

    acc = __builtin_amdgcn_wmma_f32_16x16x32_f16(
        false, af, false, bf, (short)0, acc, false, false);
  }

  if (mode & 2) {
    #pragma unroll
    for (int v = 0; v < 8; ++v) {
      int row = row0 + v + 8 * hs, col = col0 + m;
      float val = acc[v] + bias[col];
      val = val > 0.0f ? val : 0.0f;
      C[(size_t)row * HID + col] = val;
      xc[(size_t)row * TOT + xccol + col] = val;
      h16out[(size_t)row * HID + col] = (_Float16)val;
    }
  } else {
    #pragma unroll
    for (int v = 0; v < 8; ++v)
      C[(size_t)(row0 + v + 8 * hs) * HID + col0 + m] = acc[v];
  }
}

// ---------------------------------------------------------------------------
// 6) pool scorer transform-first: t[n][r] = xc[n].pool_W[r], roott[n]=xc[n].root
__global__ void k_pool_t(const float* __restrict__ xc, const float* __restrict__ poolW,
                         const float* __restrict__ poolRoot,
                         float* __restrict__ t, float* __restrict__ roott) {
  int gw   = (blockIdx.x * blockDim.x + threadIdx.x) >> 5;
  int lane = threadIdx.x & 31;
  if (gw >= NN) return;
  float xr[24];
  const float* row = xc + (size_t)gw * TOT;
  #pragma unroll
  for (int j = 0; j < 24; ++j) xr[j] = row[lane + j * 32];
  for (int r = 0; r < 9; ++r) {
    const float* w = (r < 8) ? (poolW + (size_t)r * TOT) : poolRoot;
    float p = 0.0f;
    #pragma unroll
    for (int j = 0; j < 24; ++j) p += xr[j] * w[lane + j * 32];
    #pragma unroll
    for (int off = 16; off; off >>= 1) p += __shfl_xor(p, off, 32);
    if (lane == 0) { if (r < 8) t[(size_t)gw * 8 + r] = p; else roott[gw] = p; }
  }
}

// 7) scalar edge aggregation of scores
__global__ void k_pscat(const int* __restrict__ src, const int* __restrict__ dst,
                        const int* __restrict__ et, const int* __restrict__ cnt,
                        const float* __restrict__ t, float* __restrict__ sagg) {
  int e = blockIdx.x * blockDim.x + threadIdx.x;
  if (e >= EE) return;
  int r = et[e], d = dst[e];
  float w = 1.0f / fmaxf((float)cnt[d * RR + r], 1.0f);
  atomicAdd(&sagg[d], w * t[(size_t)src[e] * 8 + r]);
}

// 8) score = tanh(agg + root + bias)
__global__ void k_score(const float* __restrict__ sagg, const float* __restrict__ roott,
                        const float* __restrict__ pb, float* __restrict__ score) {
  int n = blockIdx.x * blockDim.x + threadIdx.x;
  if (n < NN) score[n] = tanhf(sagg[n] + roott[n] + pb[0]);
}

// 9) per-graph top-k (bitonic sort 512) + weighted mean readout
__global__ void k_topk(const float* __restrict__ score, const float* __restrict__ xc,
                       float* __restrict__ readout) {
  __shared__ float sv[512];
  __shared__ int   si[512];
  int g = blockIdx.x, tid = threadIdx.x;
  for (int i = tid; i < 512; i += 256) {
    sv[i] = (i < NPG) ? score[g * NPG + i] : -1e30f;
    si[i] = i;
  }
  for (int k = 2; k <= 512; k <<= 1)
    for (int j = k >> 1; j > 0; j >>= 1) {
      __syncthreads();
      for (int i = tid; i < 512; i += 256) {
        int ixj = i ^ j;
        if (ixj > i) {
          bool up = ((i & k) == 0);          // descending overall
          float a = sv[i], b = sv[ixj];
          if (up ? (a < b) : (a > b)) {
            sv[i] = b; sv[ixj] = a;
            int ti = si[i]; si[i] = si[ixj]; si[ixj] = ti;
          }
        }
      }
    }
  __syncthreads();
  for (int f = tid; f < TOT; f += 256) {
    float acc = 0.0f;
    for (int j = 0; j < KK; ++j)
      acc += sv[j] * xc[(size_t)(g * NPG + si[j]) * TOT + f];
    readout[(size_t)g * TOT + f] = acc * (1.0f / (float)KK);
  }
}

// 10) fc1 + relu
__global__ void k_fc1(const float* __restrict__ ro, const float* __restrict__ W,
                      const float* __restrict__ b, float* __restrict__ h1) {
  int idx = blockIdx.x * blockDim.x + threadIdx.x;
  if (idx >= GG * 256) return;
  int g = idx >> 8, o = idx & 255;
  const float* r = ro + (size_t)g * TOT;
  const float* w = W + (size_t)o * TOT;
  float a = b[o];
  for (int i = 0; i < TOT; ++i) a += r[i] * w[i];
  h1[idx] = fmaxf(a, 0.0f);
}

// 11) single-block LSTM over 40 steps (gate order i,f,g,o)
__global__ void k_lstm(const float* __restrict__ h1, const float* __restrict__ Wih,
                       const float* __restrict__ Whh, const float* __restrict__ bih,
                       const float* __restrict__ bhh, float* __restrict__ hlast) {
  __shared__ float hsm[LSTM_H], csm[LSTM_H], gsm[4 * LSTM_H];
  int tid = threadIdx.x;
  if (tid < LSTM_H) { hsm[tid] = 0.0f; csm[tid] = 0.0f; }
  __syncthreads();
  for (int tstep = 0; tstep < GG; ++tstep) {
    const float* xt = h1 + (size_t)tstep * LSTM_H;
    const float* wi = Wih + (size_t)tid * LSTM_H;
    const float* wh = Whh + (size_t)tid * LSTM_H;
    float a = bih[tid] + bhh[tid];
    for (int i = 0; i < LSTM_H; ++i) a += xt[i] * wi[i] + hsm[i] * wh[i];
    gsm[tid] = a;
    __syncthreads();
    if (tid < LSTM_H) {
      float ig = 1.0f / (1.0f + expf(-gsm[tid]));
      float fg = 1.0f / (1.0f + expf(-gsm[LSTM_H + tid]));
      float gg = tanhf(gsm[2 * LSTM_H + tid]);
      float og = 1.0f / (1.0f + expf(-gsm[3 * LSTM_H + tid]));
      float c  = fg * csm[tid] + ig * gg;
      csm[tid] = c;
      hsm[tid] = og * tanhf(c);
    }
    __syncthreads();
  }
  if (tid < LSTM_H) hlast[tid] = hsm[tid];
}

// 12) fc2 + log_softmax -> d_out[2]
__global__ void k_head(const float* __restrict__ hlast, const float* __restrict__ W,
                       const float* __restrict__ b, float* __restrict__ out) {
  __shared__ float lg[2];
  int tid = threadIdx.x, w = tid >> 5, lane = tid & 31;
  if (tid < 64) {
    const float* wr = W + (size_t)w * LSTM_H;
    float p = 0.0f;
    for (int j = lane; j < LSTM_H; j += 32) p += hlast[j] * wr[j];
    #pragma unroll
    for (int off = 16; off; off >>= 1) p += __shfl_xor(p, off, 32);
    if (lane == 0) lg[w] = p + b[w];
  }
  __syncthreads();
  if (tid == 0) {
    float a = lg[0], c = lg[1];
    float mx = fmaxf(a, c);
    float ls = mx + logf(expf(a - mx) + expf(c - mx));
    out[0] = a - ls;
    out[1] = c - ls;
  }
}

// ---------------------------------------------------------------------------
extern "C" void kernel_launch(void* const* d_in, const int* in_sizes, int n_in,
                              void* d_out, int out_size, void* d_ws, size_t ws_size,
                              hipStream_t stream) {
  (void)in_sizes; (void)n_in; (void)out_size; (void)ws_size;
  const float* x        = (const float*)d_in[0];
  const int*   ei       = (const int*)d_in[1];
  const int*   src      = ei;
  const int*   dstp     = ei + EE;
  const int*   et       = (const int*)d_in[2];
  // d_in[3] = batch (implicit: contiguous equal-size graphs)
  const float* convW    = (const float*)d_in[4];
  const float* convRoot = (const float*)d_in[5];
  const float* convBias = (const float*)d_in[6];
  const float* poolW    = (const float*)d_in[7];
  const float* poolRoot = (const float*)d_in[8];
  const float* poolBias = (const float*)d_in[9];
  const float* fc1W     = (const float*)d_in[10];
  const float* fc1b     = (const float*)d_in[11];
  const float* Wih      = (const float*)d_in[12];
  const float* Whh      = (const float*)d_in[13];
  const float* bih      = (const float*)d_in[14];
  const float* bhh      = (const float*)d_in[15];
  const float* fc2W     = (const float*)d_in[16];
  const float* fc2b     = (const float*)d_in[17];

  // workspace carve-out (~160 MB)
  char* p = (char*)d_ws;
  auto take = [&](size_t bytes) -> char* {
    char* c = p;
    p += (bytes + 255) & ~(size_t)255;
    return c;
  };
  float*    buf   = (float*)take((size_t)NN * HID * 4);
  float*    hA    = (float*)take((size_t)NN * HID * 4);
  float*    hB    = (float*)take((size_t)NN * HID * 4);
  float*    xc    = (float*)take((size_t)NN * TOT * 4);
  _Float16* Wt    = (_Float16*)take((size_t)3 * 9 * 65536 * 2);
  _Float16* x16   = (_Float16*)take((size_t)NN * HID * 2);
  _Float16* h16a  = (_Float16*)take((size_t)NN * HID * 2);
  _Float16* h16b  = (_Float16*)take((size_t)NN * HID * 2);
  _Float16* buf16 = (_Float16*)take((size_t)NN * HID * 2);
  int*      cnt   = (int*)take((size_t)NN * RR * 4);
  float*    t     = (float*)take((size_t)NN * RR * 4);
  float*    roott = (float*)take((size_t)NN * 4);
  float*    sagg  = (float*)take((size_t)NN * 4);
  float*    score = (float*)take((size_t)NN * 4);
  float*    ro    = (float*)take((size_t)GG * TOT * 4);
  float*    h1    = (float*)take((size_t)GG * 256 * 4);
  float*    hlast = (float*)take((size_t)256 * 4);

  hipMemsetAsync(cnt, 0, (size_t)NN * RR * 4, stream);
  hipMemsetAsync(sagg, 0, (size_t)NN * 4, stream);
  k_count<<<(EE + 255) / 256, 256, 0, stream>>>(dstp, et, cnt);
  k_wcvt<<<(3 * 9 * 65536 + 255) / 256, 256, 0, stream>>>(convW, convRoot, Wt);
  k_cvt<<<(NN * HID / 4 + 255) / 256, 256, 0, stream>>>(x, x16, NN * HID);

  // --- 3 stacked RGCN layers: aggregate-per-relation then WMMA GEMM ---
  const float* hin = x;
  float*    houts[3]  = { hA, hB, hA };   // f32 ping-pong for scatter input
  _Float16* h16o[3]   = { h16a, h16b, h16a };
  for (int l = 0; l < 3; ++l) {
    float*    hout    = houts[l];
    _Float16* h16out  = h16o[l];
    const _Float16* hrt16 = (l == 0) ? x16 : h16o[l - 1];
    for (int r = 0; r < 9; ++r) {
      const _Float16* A16;
      if (r < 8) {
        hipMemsetAsync(buf, 0, (size_t)NN * HID * 4, stream);
        k_scatter<<<EE / 8, 256, 0, stream>>>(src, dstp, et, cnt, hin, buf, r);
        k_cvt<<<(NN * HID / 4 + 255) / 256, 256, 0, stream>>>(buf, buf16, NN * HID);
        A16 = buf16;
      } else {
        A16 = hrt16;                      // root transform on layer input (f16)
      }
      int mode = (r == 0 ? 1 : 0) | (r == 8 ? 2 : 0);
      k_gemm16<<<NN / 16, 512, 0, stream>>>(
          A16, Wt + (size_t)(l * 9 + r) * 65536, hout,
          convBias + (size_t)l * HID, xc, l * HID, h16out, mode);
    }
    hin = hout;
  }

  // --- SAGPool scorer (output dim 1 -> transform-first) ---
  k_pool_t<<<NN / 8, 256, 0, stream>>>(xc, poolW, poolRoot, t, roott);
  k_pscat<<<(EE + 255) / 256, 256, 0, stream>>>(src, dstp, et, cnt, t, sagg);
  k_score<<<(NN + 255) / 256, 256, 0, stream>>>(sagg, roott, poolBias, score);
  k_topk<<<GG, 256, 0, stream>>>(score, xc, ro);

  // --- head ---
  k_fc1<<<(GG * 256 + 255) / 256, 256, 0, stream>>>(ro, fc1W, fc1b, h1);
  k_lstm<<<1, 1024, 0, stream>>>(h1, Wih, Whh, bih, bhh, hlast);
  k_head<<<1, 64, 0, stream>>>(hlast, fc2W, fc2b, (float*)d_out);
}
